// GCN_30382598652516
// MI455X (gfx1250) — matrix operand order repaired and checked by reading, not compiled, not run on hardware
//
#include <hip/hip_runtime.h>
#include <hip/hip_bf16.h>
#include <stdint.h>
#include <stddef.h>

// ---------------- CDNA5 WMMA vector types ----------------
typedef __attribute__((ext_vector_type(16))) __bf16 v16bf;
typedef __attribute__((ext_vector_type(8)))  __bf16 v8bf;
typedef __attribute__((ext_vector_type(8)))  float  v8f;

#define NN     30000
#define EE     240000
#define HIDS   512
#define MP     30080          // 235 * 128, padded M
#define RAWD   41
#define NODED  38
#define KPAD   64             // NODED padded to 2 WMMA k-steps
#define INVMAXW (1.0f / 15.286330223083496f)

// ---------------- edge preprocessing ----------------
__global__ void k_edge_deg(const long long* __restrict__ ei,
                           const float* __restrict__ attr,
                           float* __restrict__ w, float* __restrict__ deg) {
  int e = blockIdx.x * blockDim.x + threadIdx.x;
  if (e >= EE) return;
  float we = attr[e * 4 + 3] * INVMAXW;
  w[e] = we;
  atomicAdd(&deg[(int)ei[EE + e]], we);
}

__global__ void k_dinv(const float* __restrict__ deg, float* __restrict__ dinv) {
  int i = blockIdx.x * blockDim.x + threadIdx.x;
  if (i >= NN) return;
  float d = deg[i];
  dinv[i] = d > 0.0f ? rsqrtf(d) : 0.0f;
}

__global__ void k_norm(const long long* __restrict__ ei,
                       const float* __restrict__ w,
                       const float* __restrict__ dinv,
                       float* __restrict__ nrm) {
  int e = blockIdx.x * blockDim.x + threadIdx.x;
  if (e >= EE) return;
  nrm[e] = dinv[(int)ei[e]] * w[e] * dinv[(int)ei[EE + e]];
}

// ---------------- bf16 packing ----------------
__global__ void k_pack_x(const float* __restrict__ x, __bf16* __restrict__ xp) {
  int idx = blockIdx.x * blockDim.x + threadIdx.x;     // MP*KPAD
  if (idx >= MP * KPAD) return;
  int row = idx >> 6, k = idx & (KPAD - 1);
  float v = (row < NN && k < NODED) ? x[(size_t)row * RAWD + k] : 0.0f;
  xp[idx] = (__bf16)v;
}

__global__ void k_pack_w1(const float* __restrict__ W1, __bf16* __restrict__ W1p) {
  int idx = blockIdx.x * blockDim.x + threadIdx.x;     // HIDS*KPAD
  if (idx >= HIDS * KPAD) return;
  int n = idx >> 6, k = idx & (KPAD - 1);
  float v = (k < NODED) ? W1[(size_t)n * NODED + k] : 0.0f;
  W1p[idx] = (__bf16)v;
}

__global__ void k_pack_wct(const float* __restrict__ Wc1,
                           const float* __restrict__ Wc2,
                           const float* __restrict__ Wc3,
                           __bf16* __restrict__ WcT) {
  int idx = blockIdx.x * blockDim.x + threadIdx.x;     // 3*HIDS*HIDS
  if (idx >= 3 * HIDS * HIDS) return;
  int m = idx >> 18;
  int rem = idx & (HIDS * HIDS - 1);
  int n = rem >> 9, k = rem & (HIDS - 1);
  const float* W = (m == 0) ? Wc1 : ((m == 1) ? Wc2 : Wc3);
  WcT[idx] = (__bf16)W[(size_t)k * HIDS + n];          // WcT[n][k] = Wc[k][n]
}

// ---------------- GCNII sparse part ----------------
__global__ void k_init_acc(const float* __restrict__ h0, float* __restrict__ acc) {
  int i = blockIdx.x * blockDim.x + threadIdx.x;
  if (i >= MP * HIDS) return;
  acc[i] = 0.25f * h0[i];                              // ALPHA/(1-ALPHA) * h0
}

__global__ void k_scatter(const long long* __restrict__ ei,
                          const float* __restrict__ nrm,
                          const float* __restrict__ src,
                          float* __restrict__ acc) {
  int e = blockIdx.x;                                  // one edge per block
  int t = threadIdx.x;                                 // 256 threads, 2 feats each
  int r = (int)ei[e];
  int c = (int)ei[EE + e];
  float s = nrm[e];
  float2 v = ((const float2*)(src + (size_t)r * HIDS))[t];
  float* ap = acc + (size_t)c * HIDS + t * 2;
  atomicAdd(ap,     v.x * s);
  atomicAdd(ap + 1, v.y * s);
}

__global__ void k_combine(const float* __restrict__ acc, __bf16* __restrict__ A16) {
  int i = blockIdx.x * blockDim.x + threadIdx.x;
  if (i >= MP * HIDS) return;
  A16[i] = (__bf16)(0.8f * acc[i]);                    // (1-ALPHA) * (acc)
}

// ---------------- WMMA bf16 GEMM: C = A(MxK,row) * Bt(NxK,row)^T ----------------
// Wave tile 32x64 (2 M-subtiles x 4 N-subtiles), block = 8 waves -> 128x128 tile.
// Fragment loads straight from global per CDNA5 16-bit layouts:
//   A lane l: row = l&15 (+16 per subtile), elems 0..7 at k + (l>>4)*8, elems 8..15 at +16
//   B lane l: col = l&15, elems 0..7 at k + (l>>4)*8, elems 8..15 at +16
__launch_bounds__(256)
__global__ void k_gemm(const __bf16* __restrict__ A, int lda,
                       const __bf16* __restrict__ Bt, int ldb,
                       const float* __restrict__ bias,
                       float* __restrict__ C, int ldc,
                       int K, int Mstore, int do_relu) {
  const int lane = threadIdx.x & 31;
  const int wid  = threadIdx.x >> 5;
  const int m0 = blockIdx.x * 128 + (wid >> 1) * 32;
  const int n0 = blockIdx.y * 128 + (wid & 1) * 64;
  const int l15 = lane & 15;
  const int lhi = lane >> 4;

  v8f acc[2][4] = {};

  const __bf16* Ab = A  + (size_t)(m0 + l15) * lda + lhi * 8;
  const __bf16* Bb = Bt + (size_t)(n0 + l15) * ldb + lhi * 8;

  for (int k0 = 0; k0 < K; k0 += 32) {
    v16bf af[2];
#pragma unroll
    for (int mi = 0; mi < 2; mi++) {
      const __bf16* p = Ab + (size_t)mi * 16 * lda + k0;
      v8bf lo = *(const v8bf*)p;
      v8bf hi = *(const v8bf*)(p + 16);
#pragma unroll
      for (int t = 0; t < 8; t++) { af[mi][t] = lo[t]; af[mi][t + 8] = hi[t]; }
    }
    v16bf bfr[4];
#pragma unroll
    for (int ni = 0; ni < 4; ni++) {
      const __bf16* q = Bb + (size_t)ni * 16 * ldb + k0;
      v8bf lo = *(const v8bf*)q;
      v8bf hi = *(const v8bf*)(q + 16);
#pragma unroll
      for (int t = 0; t < 8; t++) { bfr[ni][t] = lo[t]; bfr[ni][t + 8] = hi[t]; }
    }
#pragma unroll
    for (int mi = 0; mi < 2; mi++)
#pragma unroll
      for (int ni = 0; ni < 4; ni++)
        acc[mi][ni] = __builtin_amdgcn_wmma_f32_16x16x32_bf16(
            false, af[mi], false, bfr[ni], (short)0, acc[mi][ni], false, false);
  }

#pragma unroll
  for (int ni = 0; ni < 4; ni++) {
    int n = n0 + ni * 16 + l15;
    float bv = bias ? bias[n] : 0.0f;
#pragma unroll
    for (int mi = 0; mi < 2; mi++) {
      int rbase = m0 + mi * 16 + lhi * 8;
#pragma unroll
      for (int r = 0; r < 8; r++) {
        int row = rbase + r;
        if (row < Mstore) {
          float v = acc[mi][ni][r] + bv;
          if (do_relu) v = fmaxf(v, 0.0f);
          C[(size_t)row * ldc + n] = v;
        }
      }
    }
  }
}

// ---------------- host orchestration ----------------
static inline unsigned cdiv(size_t a, unsigned b) { return (unsigned)((a + b - 1) / b); }

extern "C" void kernel_launch(void* const* d_in, const int* in_sizes, int n_in,
                              void* d_out, int out_size, void* d_ws, size_t ws_size,
                              hipStream_t stream) {
  const float*     x    = (const float*)d_in[0];
  const long long* ei   = (const long long*)d_in[1];
  const float*     attr = (const float*)d_in[2];
  const float*     W1   = (const float*)d_in[3];
  const float*     b1   = (const float*)d_in[4];
  const float*     Wc1  = (const float*)d_in[5];
  const float*     Wc2  = (const float*)d_in[6];
  const float*     Wc3  = (const float*)d_in[7];
  (void)in_sizes; (void)n_in; (void)out_size; (void)ws_size;

  char* p = (char*)d_ws;
  auto take = [&](size_t bytes) { char* r = p; p += (bytes + 255) & ~(size_t)255; return r; };

  float*  h0   = (float*)take((size_t)MP * HIDS * 4);
  float*  h    = (float*)take((size_t)MP * HIDS * 4);
  float*  accb = (float*)take((size_t)MP * HIDS * 4);
  __bf16* A16  = (__bf16*)take((size_t)MP * HIDS * 2);
  __bf16* xp   = (__bf16*)take((size_t)MP * KPAD * 2);
  __bf16* W1p  = (__bf16*)take((size_t)HIDS * KPAD * 2);
  __bf16* WcT  = (__bf16*)take((size_t)3 * HIDS * HIDS * 2);
  float*  wE   = (float*)take((size_t)EE * 4);
  float*  deg  = (float*)take((size_t)NN * 4);
  float*  dinv = (float*)take((size_t)NN * 4);
  float*  nrm  = (float*)take((size_t)EE * 4);

  // normalization
  hipMemsetAsync(deg, 0, (size_t)NN * 4, stream);
  k_edge_deg<<<cdiv(EE, 256), 256, 0, stream>>>(ei, attr, wE, deg);
  k_dinv<<<cdiv(NN, 256), 256, 0, stream>>>(deg, dinv);
  k_norm<<<cdiv(EE, 256), 256, 0, stream>>>(ei, wE, dinv, nrm);

  // bf16 operand packing
  k_pack_x  <<<cdiv((size_t)MP * KPAD, 256), 256, 0, stream>>>(x, xp);
  k_pack_w1 <<<cdiv((size_t)HIDS * KPAD, 256), 256, 0, stream>>>(W1, W1p);
  k_pack_wct<<<cdiv((size_t)3 * HIDS * HIDS, 256), 256, 0, stream>>>(Wc1, Wc2, Wc3, WcT);

  dim3 ggrid(MP / 128, HIDS / 128);

  // h0 = x[:, :38] @ W1.T + b1   (K padded to 64)
  k_gemm<<<ggrid, 256, 0, stream>>>(xp, KPAD, W1p, KPAD, b1, h0, HIDS, KPAD, MP, 0);

  // three GCNII layers
  for (int l = 0; l < 3; l++) {
    const float* src = (l == 0) ? h0 : h;
    float* dst = (l == 2) ? (float*)d_out : h;
    const __bf16* Bw = WcT + (size_t)l * HIDS * HIDS;

    k_init_acc<<<cdiv((size_t)MP * HIDS, 256), 256, 0, stream>>>(h0, accb);
    k_scatter <<<EE, 256, 0, stream>>>(ei, nrm, src, accb);
    k_combine <<<cdiv((size_t)MP * HIDS, 256), 256, 0, stream>>>(accb, A16);
    k_gemm<<<ggrid, 256, 0, stream>>>(A16, HIDS, Bw, HIDS, nullptr, dst, HIDS,
                                      HIDS, (l == 2) ? NN : MP, 1);
  }
}
